// CustomNet_10952166605458
// MI455X (gfx1250) — matrix operand order repaired and checked by reading, not compiled, not run on hardware
//
#include <hip/hip_runtime.h>

// ---------------------------------------------------------------------------
// MNIST-style CNN forward on MI455X (gfx1250, wave32, WMMA).
//   k0a: fc1_w fp32 -> f16 (one-time; makes k3 staging pure b128 copies)
//   k0b: w2 fp32 -> f16 B^T layout k=(dy*3+dx)*32+c (one-time; makes k2
//        weight staging pure b128 copies instead of div+scatter per block)
//   k1: conv1(1->32,3x3)+ReLU -> h1 [B][26][26][32] f16  (VALU; K=9 too small)
//   k2: conv2(32->64,3x3)+ReLU+x-pool as implicit-GEMM WMMA f16 (K=288),
//       one image per WG, A frags preloaded per tile (pipelined ds_load_b128),
//       y-pool in epilogue; pooled out in torch-flatten order f16.
//       LDS = 130 KB -> 2 WGs / 320 KB WGP (16 waves).
//   k3: fc1(9216->128)+ReLU+fc2(128->10) fused; fc1 WMMA f16 K-chunked via LDS.
// ---------------------------------------------------------------------------

typedef __attribute__((ext_vector_type(16))) _Float16 v16h;
typedef __attribute__((ext_vector_type(8)))  _Float16 v8h;
typedef __attribute__((ext_vector_type(4)))  _Float16 v4h;
typedef __attribute__((ext_vector_type(8)))  float    v8f;

__device__ __forceinline__ v8f wmma_f32_f16(v16h a, v16h b, v8f c) {
  // (neg_a, A, neg_b, B, c_mod, C, reuse_a, reuse_b)
  return __builtin_amdgcn_wmma_f32_16x16x32_f16(false, a, false, b, (short)0, c,
                                                false, false);
}

// A-matrix 16x32 f16 fragment (ISA 7.12.2): lane = (half<<4)|r supplies row M=r,
// K = {8*half .. 8*half+7} U {16+8*half .. 16+8*half+7}. `row` points at A[r][k0].
__device__ __forceinline__ v16h load_frag_A(const _Float16* row, int half) {
  v8h lo = *(const v8h*)(row + half * 8);
  v8h hi = *(const v8h*)(row + 16 + half * 8);
  v16h r;
#pragma unroll
  for (int i = 0; i < 8; ++i) { r[i] = lo[i]; r[i + 8] = hi[i]; }
  return r;
}

// B-matrix 32x16 f16 fragment built from a B^T row (K contiguous): lane
// (half,n) supplies column N=n, K = {8*half .. 8*half+7} U {16+8*half .. 16+8*half+7}. `rowN` = Bt[n]+k0.
__device__ __forceinline__ v16h load_frag_Bt(const _Float16* rowN, int half) {
  v8h lo = *(const v8h*)(rowN + half * 8);
  v8h hi = *(const v8h*)(rowN + 16 + half * 8);
  v16h r;
#pragma unroll
  for (int i = 0; i < 8; ++i) { r[i] = lo[i]; r[i + 8] = hi[i]; }
  return r;
}

// ------------------------------ kernel 0a ----------------------------------
// fc1_w fp32 -> f16, torch layout preserved. 128*9216 elems, 8 per thread.
__global__ __launch_bounds__(256) void cvt_fc1w_k(const float* __restrict__ w,
                                                  _Float16* __restrict__ o) {
  size_t i = ((size_t)blockIdx.x * 256 + threadIdx.x) * 8;
  v8h r;
#pragma unroll
  for (int e = 0; e < 8; ++e) r[e] = (_Float16)w[i + e];
  *(v8h*)(o + i) = r;
}

// ------------------------------ kernel 0b ----------------------------------
// w2 [64][32][3][3] fp32 -> f16 B^T [64][288], k = (dy*3+dx)*32 + c.
__global__ __launch_bounds__(256) void cvt_w2_k(const float* __restrict__ w2,
                                                _Float16* __restrict__ w2t) {
  int i = blockIdx.x * 256 + threadIdx.x;  // 64*288 total
  int o = i / 288, r = i % 288, c = r / 9, s = r % 9;
  w2t[o * 288 + s * 32 + c] = (_Float16)w2[i];
}

// ------------------------------ kernel 1 -----------------------------------
// One thread per output pixel, all 32 channels in registers. h1 channel-last.
__global__ __launch_bounds__(256) void conv1_relu_k(
    const float* __restrict__ x, const float* __restrict__ w1,
    const float* __restrict__ b1, _Float16* __restrict__ h1, int total) {
  __shared__ float sw[32 * 9 + 32];
  for (int i = threadIdx.x; i < 320; i += 256)
    sw[i] = (i < 288) ? w1[i] : b1[i - 288];
  __syncthreads();
  int pid = blockIdx.x * 256 + threadIdx.x;
  if (pid >= total) return;
  int b = pid / 676, p = pid % 676;
  int y = p / 26, xx = p % 26;
  const float* xim = x + (size_t)b * 784;
  float acc[32];
#pragma unroll
  for (int c = 0; c < 32; ++c) acc[c] = sw[288 + c];
#pragma unroll
  for (int dy = 0; dy < 3; ++dy)
#pragma unroll
    for (int dx = 0; dx < 3; ++dx) {
      float v = xim[(y + dy) * 28 + xx + dx];
#pragma unroll
      for (int c = 0; c < 32; ++c) acc[c] += v * sw[c * 9 + dy * 3 + dx];
    }
  _Float16* op = h1 + (size_t)pid * 32;
#pragma unroll
  for (int g = 0; g < 4; ++g) {
    v8h o;
#pragma unroll
    for (int e = 0; e < 8; ++e) o[e] = (_Float16)fmaxf(acc[g * 8 + e], 0.f);
    *(v8h*)(op + g * 8) = o;
  }
}

// ------------------------------ kernel 2 -----------------------------------
#define K2_CP 40   // padded channel dim of staged h1 (20 dwords: conflict-free)
#define K2_WP 296  // padded K dim of B^T weights (37x16B rows)
#define K2_QP 296  // padded row of x-pooled staging [c][y24*12+x12]
__global__ __launch_bounds__(256) void conv2_pool_k(
    const _Float16* __restrict__ h1, const _Float16* __restrict__ w2t,
    const float* __restrict__ b2, _Float16* __restrict__ pooled) {
  extern __shared__ char smem[];
  _Float16* sH  = (_Float16*)smem;        // [26][26][K2_CP]
  _Float16* sW  = sH + 26 * 26 * K2_CP;   // [64][K2_WP]  B^T: Wt[o][k]
  _Float16* sOx = sW + 64 * K2_WP;        // [64][K2_QP]  relu+x-pooled conv
  float*    sBias = (float*)(sOx + 64 * K2_QP);
  const int b = blockIdx.x;

  // Stage h1 image (channel-last) and pre-permuted w2t: all b128 copies.
  for (int q = threadIdx.x; q < 26 * 26 * 4; q += 256) {
    int pix = q >> 2, g = q & 3;
    *(v8h*)(sH + pix * K2_CP + g * 8) =
        *(const v8h*)(h1 + ((size_t)b * 676 + pix) * 32 + g * 8);
  }
  for (int q = threadIdx.x; q < 64 * 36; q += 256) {
    int o = q / 36, g = q % 36;
    *(v8h*)(sW + o * K2_WP + g * 8) = *(const v8h*)(w2t + o * 288 + g * 8);
  }
  if (threadIdx.x < 64) sBias[threadIdx.x] = b2[threadIdx.x];
  __syncthreads();

  const int lane = threadIdx.x & 31, w = threadIdx.x >> 5;
  const int half = lane >> 4, ln = lane & 15;
  const int nt = w & 3;   // N-tile (16 output channels)
  const int mg = w >> 2;  // M-tile parity

  // Cache all 9 B fragments for this wave's N-tile (72 VGPRs).
  v16h bf[9];
#pragma unroll
  for (int ks = 0; ks < 9; ++ks)
    bf[ks] = load_frag_Bt(sW + (nt * 16 + ln) * K2_WP + ks * 32, half);

  const float bias = sBias[nt * 16 + ln];

  // Lane's pixel coordinate; advances by 32 per tile (branchless wrap,
  // replaces per-tile division -- keeps the hot loop WMMA/DS dominated).
  int p0 = mg * 16 + ln;
  int py = p0 / 24, px = p0 % 24;

  for (int t = 0; t < 18; ++t) {
    int mt = mg + 2 * t;  // M-tile over 576 pixels
    const _Float16* base = sH + (py * 26 + px) * K2_CP;
    // Preload all 9 A fragments (distinct regs -> 18 ds_load_b128 pipelined).
    v16h af[9];
#pragma unroll
    for (int ks = 0; ks < 9; ++ks)
      af[ks] = load_frag_A(base + ((ks / 3) * 26 + (ks % 3)) * K2_CP, half);
    v8f acc = {};
#pragma unroll
    for (int ks = 0; ks < 9; ++ks) acc = wmma_f32_f16(af[ks], bf[ks], acc);

    // D element g -> pixel mt*16 + 8*half + g (8-runs never cross a 24-row),
    // channel nt*16+ln. Fuse bias+ReLU+x-direction of 2x2 maxpool:
    // max(relu(a+b), relu(c+b)) == relu(max(a,c)+b).
    int pb = mt * 16 + half * 8;
    int q0 = (pb / 24) * 12 + ((pb % 24) >> 1);
    v4h o4;
#pragma unroll
    for (int e = 0; e < 4; ++e)
      o4[e] = (_Float16)fmaxf(fmaxf(acc[2 * e], acc[2 * e + 1]) + bias, 0.f);
    *(v4h*)(sOx + (nt * 16 + ln) * K2_QP + q0) = o4;

    // p += 32  ==  (y += 1, x += 8) with wrap (data ops only, no branches).
    px += 8;
    py += 1;
    int wrap = (px >= 24) ? 1 : 0;
    px -= 24 * wrap;
    py += wrap;
  }
  __syncthreads();

  // y-direction of maxpool; write torch-flatten order [b][c*144 + y*12 + x].
  for (int idx = threadIdx.x; idx < 9216; idx += 256) {
    int c = idx / 144, p = idx % 144;
    const _Float16* r0 = sOx + c * K2_QP + (p / 12) * 2 * 12 + (p % 12);
    pooled[(size_t)b * 9216 + idx] =
        (_Float16)fmaxf((float)r0[0], (float)r0[12]);
  }
}

// ------------------------------ kernel 3 -----------------------------------
#define K3_AP 264  // padded K-chunk (33x16B rows: conflict-free)
#define K3_HP 72   // padded batch dim of h staging
__global__ __launch_bounds__(256) void fc_fused_k(
    const _Float16* __restrict__ pooled, const _Float16* __restrict__ fc1wh,
    const float* __restrict__ fc1b, const float* __restrict__ fc2w,
    const float* __restrict__ fc2b, float* __restrict__ out) {
  extern __shared__ char smem[];
  _Float16* sA  = (_Float16*)smem;        // [64][K3_AP]  activations chunk
  _Float16* sB  = sA + 64 * K3_AP;        // [128][K3_AP] fc1_w^T chunk
  _Float16* sHH = sB + 128 * K3_AP;       // [128][K3_HP] relu(fc1) [j][b]
  const int bbase = blockIdx.x * 64;
  const int lane = threadIdx.x & 31, w = threadIdx.x >> 5;
  const int half = lane >> 4, ln = lane & 15;
  const int mt = w & 3;   // M-subtile (16 batch rows)
  const int ng = w >> 2;  // N-group: ntiles ng*4 .. ng*4+3

  v8f acc[4];
#pragma unroll
  for (int j = 0; j < 4; ++j) acc[j] = (v8f){};

  for (int k0 = 0; k0 < 9216; k0 += 256) {
    __syncthreads();  // previous chunk fully consumed before overwrite
    for (int q = threadIdx.x; q < 64 * 32; q += 256) {
      int r = q >> 5, g = q & 31;
      *(v8h*)(sA + r * K3_AP + g * 8) =
          *(const v8h*)(pooled + (size_t)(bbase + r) * 9216 + k0 + g * 8);
    }
    for (int q = threadIdx.x; q < 128 * 32; q += 256) {
      int j = q >> 5, g = q & 31;
      *(v8h*)(sB + j * K3_AP + g * 8) =
          *(const v8h*)(fc1wh + (size_t)j * 9216 + k0 + g * 8);
    }
    __syncthreads();
    // Preload the 8 A fragments of this chunk (distinct regs).
    v16h af[8];
#pragma unroll
    for (int ks = 0; ks < 8; ++ks)
      af[ks] = load_frag_A(sA + (mt * 16 + ln) * K3_AP + ks * 32, half);
    // Process ks in pairs: 16 ds_load_b128 in flight, then 8 independent-chain
    // WMMAs (4 accumulators) per wait.
#pragma unroll
    for (int ks = 0; ks < 8; ks += 2) {
      v16h bfr[8];
#pragma unroll
      for (int j = 0; j < 4; ++j) {
        const _Float16* rn = sB + ((ng * 4 + j) * 16 + ln) * K3_AP;
        bfr[j]     = load_frag_Bt(rn + ks * 32, half);
        bfr[4 + j] = load_frag_Bt(rn + (ks + 1) * 32, half);
      }
#pragma unroll
      for (int j = 0; j < 4; ++j) acc[j] = wmma_f32_f16(af[ks], bfr[j], acc[j]);
#pragma unroll
      for (int j = 0; j < 4; ++j)
        acc[j] = wmma_f32_f16(af[ks + 1], bfr[4 + j], acc[j]);
    }
  }
  __syncthreads();
  // Bias + ReLU -> h staged [j][b] f16.
#pragma unroll
  for (int j = 0; j < 4; ++j) {
    int nn = (ng * 4 + j) * 16 + ln;
    float bias = fc1b[nn];
    v8h o;
#pragma unroll
    for (int g = 0; g < 8; ++g) o[g] = (_Float16)fmaxf(acc[j][g] + bias, 0.f);
    *(v8h*)(sHH + nn * K3_HP + mt * 16 + half * 8) = o;
  }
  __syncthreads();
  // fc2: 64 rows x 10 outputs, K=128 -> VALU (10 MFLOP total).
  for (int idx = threadIdx.x; idx < 640; idx += 256) {
    int bb = idx & 63, o = idx >> 6;
    float s = fc2b[o];
    for (int j = 0; j < 128; ++j)
      s += (float)sHH[j * K3_HP + bb] * fc2w[o * 128 + j];
    out[(size_t)(bbase + bb) * 10 + o] = s;
  }
}

// ------------------------------ launcher -----------------------------------
extern "C" void kernel_launch(void* const* d_in, const int* in_sizes, int n_in,
                              void* d_out, int out_size, void* d_ws,
                              size_t ws_size, hipStream_t stream) {
  const float* x    = (const float*)d_in[0];
  const float* w1   = (const float*)d_in[1];
  const float* b1   = (const float*)d_in[2];
  const float* w2   = (const float*)d_in[3];
  const float* b2   = (const float*)d_in[4];
  const float* fc1w = (const float*)d_in[5];
  const float* fc1b = (const float*)d_in[6];
  const float* fc2w = (const float*)d_in[7];
  const float* fc2b = (const float*)d_in[8];
  float* out = (float*)d_out;

  // Workspace carve: h1 f16 (177.2 MB) + pooled f16 (75.5 MB) + fc1w f16
  // (2.36 MB) + w2t f16 (36.9 KB) = 255.1 MB.
  char* ws = (char*)d_ws;
  size_t off = 0;
  _Float16* h1     = (_Float16*)(ws + off); off += (size_t)4096 * 676 * 32 * 2;
  _Float16* pooled = (_Float16*)(ws + off); off += (size_t)4096 * 9216 * 2;
  _Float16* fc1wh  = (_Float16*)(ws + off); off += (size_t)128 * 9216 * 2;
  _Float16* w2t    = (_Float16*)(ws + off);

  const size_t smem2 =
      (size_t)(26 * 26 * K2_CP + 64 * K2_WP + 64 * K2_QP) * sizeof(_Float16) +
      64 * sizeof(float);  // 130,112 B -> 2 workgroups per 320 KB WGP
  const size_t smem3 =
      (size_t)(64 * K3_AP + 128 * K3_AP + 128 * K3_HP) * sizeof(_Float16);

  (void)hipFuncSetAttribute((const void*)conv2_pool_k,
                            hipFuncAttributeMaxDynamicSharedMemorySize,
                            (int)smem2);
  (void)hipFuncSetAttribute((const void*)fc_fused_k,
                            hipFuncAttributeMaxDynamicSharedMemorySize,
                            (int)smem3);

  cvt_fc1w_k<<<(128 * 9216) / (256 * 8), 256, 0, stream>>>(fc1w, fc1wh);
  cvt_w2_k<<<(64 * 288) / 256, 256, 0, stream>>>(w2, w2t);
  const int total1 = 4096 * 676;
  conv1_relu_k<<<(total1 + 255) / 256, 256, 0, stream>>>(x, w1, b1, h1, total1);
  conv2_pool_k<<<4096, 256, smem2, stream>>>(h1, w2t, b2, pooled);
  fc_fused_k<<<64, 256, smem3, stream>>>(pooled, fc1wh, fc1b, fc2w, fc2b, out);
}
